// MultiHeadAttentionLayer_592705487326
// MI455X (gfx1250) — matrix operand order repaired and checked by reading, not compiled, not run on hardware
//
#include <hip/hip_runtime.h>
#include <hip/hip_bf16.h>
#include <math.h>

#define N_NODES  50000
#define N_EDGES  800000
#define IN_DIM   128
#define HEADS    8
#define HEAD_DIM 16
#define D_MODEL  128   // HEADS * HEAD_DIM

typedef __attribute__((ext_vector_type(16))) __bf16 v16bf;
typedef __attribute__((ext_vector_type(8)))  float  v8f;

// ---------------------------------------------------------------------------
// WMMA helpers (CDNA5 wave32, V_WMMA_F32_16X16X32_BF16)
// ---------------------------------------------------------------------------
__device__ __forceinline__ v8f wmma_bf16(v16bf a, v16bf b, v8f c) {
  // (neg_a, A, neg_b, B, c_mod, C, reuse_a, reuse_b)
  return __builtin_amdgcn_wmma_f32_16x16x32_bf16(false, a, false, b,
                                                 (short)0, c, false, false);
}

// A-matrix fragment (16x32 bf16, MxK) from a row-major f32 row.
// Layout: lanes 0-15 hold M=lane, K=0..15; lanes 16-31 hold
// M=lane-16, K=16..31; element pairs are consecutive K.
__device__ __forceinline__ v16bf load_a_frag_f32(const float* __restrict__ Arow,
                                                 int k0, int lane) {
  const int off = (lane < 16) ? 0 : 16;
  const float* p1 = Arow + k0 + off;
  const float* p2 = Arow + k0 + 8 + off;
  v16bf a;
#pragma unroll
  for (int i = 0; i < 8; ++i) a[i] = (__bf16)p1[i];
#pragma unroll
  for (int i = 0; i < 8; ++i) a[8 + i] = (__bf16)p2[i];
  return a;
}

// B-matrix fragment (32x16 bf16, KxN) from LDS holding W transposed as
// WT[n*128 + k] (so per-lane K runs are contiguous -> one 32B LDS load).
// Layout: lanes 0-15 = col n, K=0..15; lanes 16-31 = col n, K=16..31.
__device__ __forceinline__ v16bf load_b_frag_lds(const __bf16* WT, int n,
                                                 int k0, int lane) {
  const int kb = k0 + ((lane < 16) ? 0 : 16);
  return *(const v16bf*)(WT + n * 128 + kb);
}

// Cooperative load of a 128x128 f32 weight into LDS as transposed bf16:
// WT[n*128 + k] = bf16(W[k*128 + n]).  LDS writes are fully contiguous.
__device__ __forceinline__ void load_weight_transposed(
    const float* __restrict__ W, __bf16* WT, int tid, int nthreads) {
  for (int i = tid; i < 128 * 128; i += nthreads) {
    const int n = i >> 7;
    const int k = i & 127;
    WT[i] = (__bf16)W[k * 128 + n];
  }
}

// ---------------------------------------------------------------------------
// Kernel 1: node projections  Q/K/V = h @ W{q,k,v} + b   (bf16 WMMA, f32 acc)
// One wave per 16-row tile; 8 waves/block; weight staged in 32KB LDS,
// looped over the three matrices so A-fragments (h rows) load only once.
// ---------------------------------------------------------------------------
__global__ __launch_bounds__(256) void node_proj_wmma(
    const float* __restrict__ h,
    const float* __restrict__ Wq, const float* __restrict__ bq,
    const float* __restrict__ Wk, const float* __restrict__ bk,
    const float* __restrict__ Wv, const float* __restrict__ bv,
    float* __restrict__ Qo, float* __restrict__ Ko, float* __restrict__ Vo) {
  __shared__ __bf16 WT[128 * 128];
  const int tid  = threadIdx.x;
  const int wave = tid >> 5;
  const int lane = tid & 31;
  const int tile = blockIdx.x * 8 + wave;
  const bool valid = tile < (N_NODES / 16);
  const int row0 = tile * 16;
  const int m    = lane & 15;
  const int hi8  = (lane < 16) ? 0 : 8;

  v16bf afrag[4];
  if (valid) {
    const float* Arow = h + (size_t)(row0 + m) * IN_DIM;
#pragma unroll
    for (int ks = 0; ks < 4; ++ks) afrag[ks] = load_a_frag_f32(Arow, ks * 32, lane);
  }

  const float* Ws[3] = {Wq, Wk, Wv};
  const float* bs[3] = {bq, bk, bv};
  float*       Os[3] = {Qo, Ko, Vo};

  for (int mat = 0; mat < 3; ++mat) {
    __syncthreads();
    load_weight_transposed(Ws[mat], WT, tid, 256);
    __syncthreads();
    if (!valid) continue;

    v8f acc[8] = {};
#pragma unroll
    for (int ks = 0; ks < 4; ++ks) {
#pragma unroll
      for (int t = 0; t < 8; ++t) {
        v16bf b = load_b_frag_lds(WT, t * 16 + m, ks * 32, lane);
        acc[t] = wmma_bf16(afrag[ks], b, acc[t]);
      }
    }
    // C layout: VGPR r -> row (r + hi8), col = t*16 + m.  Coalesced 64B rows.
    float* out = Os[mat];
    const float* bias = bs[mat];
#pragma unroll
    for (int t = 0; t < 8; ++t) {
      const int col = t * 16 + m;
      const float bb = bias[col];
#pragma unroll
      for (int r = 0; r < 8; ++r)
        out[(size_t)(row0 + r + hi8) * D_MODEL + col] = acc[t][r] + bb;
    }
  }
}

// ---------------------------------------------------------------------------
// Kernel 2: fused edge kernel.
//   proj_e = e @ We + be        (WMMA, stays in accumulator registers)
//   score  = K[src]*Q[dst]/4 * proj_e   -> e_out
//   s      = exp(clip(sum_head score))  (shfl_xor reduce: head == 16-lane half)
//   atomics: wV[dst] += V[src]*s ; z[dst,h] += s
// ---------------------------------------------------------------------------
__global__ __launch_bounds__(256) void edge_attn_wmma(
    const float* __restrict__ e, const int* __restrict__ src,
    const int* __restrict__ dst,
    const float* __restrict__ We, const float* __restrict__ be,
    const float* __restrict__ Qn, const float* __restrict__ Kn,
    const float* __restrict__ Vn,
    float* __restrict__ e_out, float* __restrict__ wV, float* __restrict__ z) {
  __shared__ __bf16 WT[128 * 128];
  const int tid = threadIdx.x;
  load_weight_transposed(We, WT, tid, 256);
  __syncthreads();

  const int wave = tid >> 5;
  const int lane = tid & 31;
  const int tile = blockIdx.x * 8 + wave;   // 50000 tiles / 8 waves -> exact
  const int row0 = tile * 16;
  const int m    = lane & 15;
  const int hi8  = (lane < 16) ? 0 : 8;

  // ---- GEMM: proj_e tile (16 edges x 128), kept in registers ----
  v8f acc[8] = {};
  {
    const float* Arow = e + (size_t)(row0 + m) * IN_DIM;
#pragma unroll
    for (int ks = 0; ks < 4; ++ks) {
      v16bf a = load_a_frag_f32(Arow, ks * 32, lane);
#pragma unroll
      for (int t = 0; t < 8; ++t) {
        v16bf b = load_b_frag_lds(WT, t * 16 + m, ks * 32, lane);
        acc[t] = wmma_bf16(a, b, acc[t]);
      }
    }
  }

  int srcs[8], dsts[8];
#pragma unroll
  for (int r = 0; r < 8; ++r) {
    srcs[r] = src[row0 + r + hi8];
    dsts[r] = dst[row0 + r + hi8];
  }

  // head t == col-tile t; its 16 columns live in one 16-lane half -> reduce
  // with shfl_xor masks 1/2/4/8 (never crosses the 16-lane boundary, wave32).
#pragma unroll
  for (int t = 0; t < 8; ++t) {
    const int col = t * 16 + m;
    const float bias = be[col];
#pragma unroll
    for (int r = 0; r < 8; ++r) {
      const float pe = acc[t][r] + bias;
      const float kq = Kn[(size_t)srcs[r] * D_MODEL + col] *
                       Qn[(size_t)dsts[r] * D_MODEL + col] * 0.25f;  // 1/sqrt(16)
      const float score = kq * pe;
      e_out[(size_t)(row0 + r + hi8) * D_MODEL + col] = score;  // coalesced 64B

      float sum = score;
      sum += __shfl_xor(sum, 1);
      sum += __shfl_xor(sum, 2);
      sum += __shfl_xor(sum, 4);
      sum += __shfl_xor(sum, 8);
      const float s = expf(fminf(fmaxf(sum, -5.0f), 5.0f));

      if (m == 0) atomicAdd(&z[(size_t)dsts[r] * HEADS + t], s);
      const float v = Vn[(size_t)srcs[r] * D_MODEL + col];
      atomicAdd(&wV[(size_t)dsts[r] * D_MODEL + col], v * s);  // L2-resident
    }
  }
}

// ---------------------------------------------------------------------------
// Kernel 3/4: init + normalize
// ---------------------------------------------------------------------------
__global__ void zero_kernel(float* __restrict__ p, int n) {
  int i = blockIdx.x * blockDim.x + threadIdx.x;
  const int stride = gridDim.x * blockDim.x;
  for (; i < n; i += stride) p[i] = 0.0f;
}

__global__ void normalize_kernel(float* __restrict__ hout,
                                 const float* __restrict__ z) {
  const int i = blockIdx.x * blockDim.x + threadIdx.x;
  if (i < N_NODES * D_MODEL) {
    const int n  = i >> 7;
    const int hd = (i >> 4) & 7;
    hout[i] = hout[i] / (z[n * HEADS + hd] + 1e-6f);
  }
}

// ---------------------------------------------------------------------------
extern "C" void kernel_launch(void* const* d_in, const int* in_sizes, int n_in,
                              void* d_out, int out_size, void* d_ws,
                              size_t ws_size, hipStream_t stream) {
  (void)in_sizes; (void)n_in; (void)out_size; (void)ws_size;

  const float* h  = (const float*)d_in[0];
  const float* e  = (const float*)d_in[1];
  const int*   src = (const int*)d_in[2];
  const int*   dst = (const int*)d_in[3];
  const float* Wq = (const float*)d_in[4];
  const float* bq = (const float*)d_in[5];
  const float* Wk = (const float*)d_in[6];
  const float* bk = (const float*)d_in[7];
  const float* Wv = (const float*)d_in[8];
  const float* bv = (const float*)d_in[9];
  const float* We = (const float*)d_in[10];
  const float* be = (const float*)d_in[11];

  float* h_out = (float*)d_out;                         // [N, H*D] = wV accum
  float* e_out = h_out + (size_t)N_NODES * D_MODEL;     // [E, H*D]

  float* Qn = (float*)d_ws;                             // 25.6 MB
  float* Kn = Qn + (size_t)N_NODES * D_MODEL;           // 25.6 MB
  float* Vn = Kn + (size_t)N_NODES * D_MODEL;           // 25.6 MB
  float* z  = Vn + (size_t)N_NODES * D_MODEL;           // 1.6 MB

  zero_kernel<<<1024, 256, 0, stream>>>(h_out, N_NODES * D_MODEL);
  zero_kernel<<<256, 256, 0, stream>>>(z, N_NODES * HEADS);

  node_proj_wmma<<<(N_NODES / 16 + 7) / 8, 256, 0, stream>>>(
      h, Wq, bq, Wk, bk, Wv, bv, Qn, Kn, Vn);

  edge_attn_wmma<<<N_EDGES / 16 / 8, 256, 0, stream>>>(
      e, src, dst, We, be, Qn, Kn, Vn, e_out, h_out, z);

  normalize_kernel<<<(N_NODES * D_MODEL + 255) / 256, 256, 0, stream>>>(h_out, z);
}
